// GATED_MHAtt_79173427134839
// MI455X (gfx1250) — matrix operand order repaired and checked by reading, not compiled, not run on hardware
//
#include <hip/hip_runtime.h>
#include <hip/hip_bf16.h>

// ---------------------------------------------------------------------------
// Gated MH attention with geometric bias, MI455X (gfx1250, wave32, WMMA).
// Bandwidth-bound on region_rel (402 MB): geometry einsum is fused into the
// attention kernel and each region_rel tile is shared by all 8 heads of a
// workgroup. All GEMMs run on v_wmma_f32_16x16x32_bf16.
// This revision: software-pipelined K-loops (double-buffered fragments so
// loads overlap WMMA instead of s_wait_loadcnt 0 stalls), LDS bank-conflict
// padding (A-tile rows 1040B, P rows 80B, geo stride 9), and khT B-fragments
// hoisted above the geometry phase so their latency hides under WMMAs.
// ---------------------------------------------------------------------------

#define B_   16
#define S_   256
#define H_   512
#define NH_  8
#define DK_  64
#define GF_  96

#define AROW  (H_ + 8)     // padded LDS row stride (bf16) -> 4-bank skew/row
#define PROW  40           // padded P-tile row stride (bf16), 80B
#define GSTR  9            // padded geo head stride (f32)

typedef __bf16 bf16;
typedef __attribute__((ext_vector_type(16))) __bf16 v16bf;
typedef __attribute__((ext_vector_type(8)))  __bf16 v8bf;
typedef __attribute__((ext_vector_type(8)))  float  v8f;

static __device__ inline v8f v8f_zero() {
    v8f z = {0.f, 0.f, 0.f, 0.f, 0.f, 0.f, 0.f, 0.f};
    return z;
}

static __device__ inline v8f wmma_bf16(v16bf a, v16bf b, v8f c) {
    // D = A(16x32 bf16) * B(32x16 bf16) + C(16x16 f32)
    return __builtin_amdgcn_wmma_f32_16x16x32_bf16(
        /*neg_a=*/false, a, /*neg_b=*/false, b,
        /*c_mod=*/(short)0, c, /*reuse_a=*/false, /*reuse_b=*/false);
}

// A-fragment loader (16-bit A 16x32 layout): lane l holds row M=l&15;
// element i<8 -> K = 8*half + i ; element i>=8 -> K = 16 + 8*half + (i-8).
// rowp points at K=0 of this lane's row (bf16, contiguous in K).
static __device__ inline v16bf a_frag_bf16(const bf16* rowp, int lane) {
    const int half = (lane >> 4) & 1;
    v8bf lo = *(const v8bf*)(rowp + 8 * half);
    v8bf hi = *(const v8bf*)(rowp + 16 + 8 * half);
    v16bf a;
#pragma unroll
    for (int i = 0; i < 8; ++i) { a[i] = lo[i]; a[i + 8] = hi[i]; }
    return a;
}

static __device__ inline float fast_sigmoid(float x) {
    return 1.0f / (1.0f + __expf(-x));
}

// ---------------------------------------------------------------------------
// Kernel 1: weight repack. WT[n][k] = W[n][k] in bf16 (so WMMA B-fragments
// are contiguous 32B loads). WgT[n][f] = Wg[n][f] for n<8, else 0.
// ---------------------------------------------------------------------------
__global__ __launch_bounds__(256) void cvt_kernel(
    const float* __restrict__ Wk, const float* __restrict__ Wq,
    const float* __restrict__ Wm, const float* __restrict__ Wg,
    bf16* __restrict__ WkT, bf16* __restrict__ WqT,
    bf16* __restrict__ WmT, bf16* __restrict__ WgT)
{
    const int i = blockIdx.x * 256 + threadIdx.x;
    if (i < H_ * H_) {
        const int k = i >> 9;
        const int n = i & (H_ - 1);
        WkT[n * H_ + k] = (bf16)Wk[n * H_ + k];
        WqT[n * H_ + k] = (bf16)Wq[n * H_ + k];
        WmT[n * H_ + k] = (bf16)Wm[n * H_ + k];
    }
    if (i < GF_ * 16) {
        const int f = i >> 4;
        const int n = i & 15;
        WgT[n * GF_ + f] = (n < NH_) ? (bf16)Wg[n * GF_ + f] : (bf16)0.0f;
    }
}

// ---------------------------------------------------------------------------
// Kernel 2: projection + gating.
//   y = X @ W.T + b ; gate = sigmoid(y.wg + bg + c.wc + bc) ; ra = sigmoid(rabs)
// mode 0 (k-path): out_plain = y stored TRANSPOSED (vh, [B,NH,DK,S]) ; out_gated = ra*gate*y
//                  stored as khT [B,NH,S,DK].
// mode 1 (q-path): out_gated = ra*gate*y as qh [B,NH,S,DK].
// Block = 16 rows of (B*S), 8 waves x 4 N-tiles, pipelined WMMA over K=512.
// ---------------------------------------------------------------------------
__global__ __launch_bounds__(256) void proj_kernel(
    const float* __restrict__ X,  const float* __restrict__ Cc,
    const bf16*  __restrict__ WT, const float* __restrict__ bias,
    const float* __restrict__ wg, const float* __restrict__ bg1,
    const float* __restrict__ wc, const float* __restrict__ bc1,
    const float* __restrict__ rabs,
    bf16* __restrict__ out_gated, bf16* __restrict__ out_plain,
    int mode)
{
    __shared__ __align__(32) bf16  Abuf[16 * AROW]; // ~16.3 KB, bank-skewed rows
    __shared__ float Obuf[16 * H_];                 // 32 KB (fp32 projection)
    __shared__ float redK[16], redC[16];

    const int t    = threadIdx.x;
    const int wave = t >> 5;
    const int lane = t & 31;
    const int half = lane >> 4;
    const int l16  = lane & 15;
    const int r0   = blockIdx.x * 16;

    if (t < 16) { redK[t] = 0.f; redC[t] = 0.f; }
    __syncthreads();

    // Stage input rows into LDS (fp32 -> bf16) and accumulate c . wc per row.
    {
        const int row = t >> 4;
        const int c0  = (t & 15) * 32;
        const float* xp = X  + (size_t)(r0 + row) * H_ + c0;
        const float* cp = Cc + (size_t)(r0 + row) * H_ + c0;
        float part = 0.f;
#pragma unroll 8
        for (int i = 0; i < 32; ++i) {
            Abuf[row * AROW + c0 + i] = (bf16)xp[i];
            part += cp[i] * wc[c0 + i];
        }
        atomicAdd(&redC[row], part);    // ds_add_f32
    }
    __syncthreads();

    // WMMA GEMM: each wave owns 4 N-tiles of 16 columns.
    // Double-buffered K-loop: loads for step k+32 issue before wmma(k), so the
    // scheduler keeps them in flight under the matrix op.
    const bf16* arow = Abuf + l16 * AROW;
#pragma unroll
    for (int nt = 0; nt < 4; ++nt) {
        const int n0 = (wave + nt * 8) * 16;
        v8f acc = v8f_zero();
        v16bf a_cur = a_frag_bf16(arow, lane);
        v16bf b_cur = a_frag_bf16(WT + (size_t)(n0 + l16) * H_, lane);
#pragma unroll
        for (int k0 = 0; k0 < H_; k0 += 32) {
            v16bf a_nxt = a_cur, b_nxt = b_cur;
            if (k0 + 32 < H_) {
                a_nxt = a_frag_bf16(arow + k0 + 32, lane);
                b_nxt = a_frag_bf16(WT + (size_t)(n0 + l16) * H_ + k0 + 32, lane);
            }
            acc = wmma_bf16(a_cur, b_cur, acc);
            a_cur = a_nxt; b_cur = b_nxt;
        }
        const int col = n0 + l16;
        const float bv = bias[col];
#pragma unroll
        for (int r = 0; r < 8; ++r)
            Obuf[(r + 8 * half) * H_ + col] = acc[r] + bv;
    }
    __syncthreads();

    // Row-wide gate dot: y . wg
    {
        const int row = t >> 4;
        const int c0  = (t & 15) * 32;
        float part = 0.f;
#pragma unroll 8
        for (int i = 0; i < 32; ++i) part += Obuf[row * H_ + c0 + i] * wg[c0 + i];
        atomicAdd(&redK[row], part);
    }
    __syncthreads();

    // Gate, scale by sigmoid(region_abs), split heads, store bf16.
    {
        const int row = t >> 4;
        const int c0  = (t & 15) * 32;
        const int g   = r0 + row;
        const int bb  = g >> 8;          // / S_
        const int ss  = g & (S_ - 1);
        const float gate = fast_sigmoid(redK[row] + bg1[0] + redC[row] + bc1[0]);
        const float ra   = fast_sigmoid(rabs[g]);
        const float sc   = gate * ra;
#pragma unroll 8
        for (int i = 0; i < 32; ++i) {
            const int col = c0 + i;
            const float val = Obuf[row * H_ + col];
            const int hh = col >> 6;     // / DK_
            const int d  = col & (DK_ - 1);
            if (mode == 0) {
                // value = ungated projection, stored transposed; key = gated
                out_plain[((size_t)(bb * NH_ + hh) * DK_ + d) * S_ + ss] = (bf16)val;
                out_gated[((size_t)(bb * NH_ + hh) * S_ + ss) * DK_ + d] = (bf16)(sc * val);
            } else {
                out_gated[((size_t)(bb * NH_ + hh) * S_ + ss) * DK_ + d] = (bf16)(sc * val);
            }
        }
    }
}

// ---------------------------------------------------------------------------
// Kernel 3: fused geometry-bias + flash attention.
// Block = (batch, 16-query tile); wave w == head w. Loop over 32-key slabs:
//   * khT B-fragments loaded FIRST (latency hidden under geometry WMMAs)
//   * all 8 waves cooperatively compute geometry bias for ALL heads:
//       A = 16 (q,k)-pairs x 96 feats (bf16), B = WgT (heads in cols 0..7),
//       3 chained WMMAs; log(max(x+bg,1e-6)) -> LDS   (region_rel read ONCE)
//   * per-wave QK^T (4 WMMAs), masked online softmax (shfl_xor in 16-lane
//     column groups), P -> LDS -> A-frag, att @ V (4 WMMAs, fp32 accum)
// ---------------------------------------------------------------------------
__global__ __launch_bounds__(256) void attn_kernel(
    const bf16* __restrict__ qh, const bf16* __restrict__ khT,
    const bf16* __restrict__ vh,
    const float* __restrict__ rr, const bf16* __restrict__ WgT,
    const float* __restrict__ bg, const unsigned char* __restrict__ mask,
    bf16* __restrict__ attout)
{
    __shared__ float geo[16 * 32 * GSTR];              // [qi][k 0..31][head*], 18 KB
    __shared__ __align__(32) bf16 Pbuf[NH_ * 16 * PROW]; // per-head P, 80B rows, 10 KB
    __shared__ float msk[S_];

    const int t    = threadIdx.x;
    const int wave = t >> 5;          // == head
    const int lane = t & 31;
    const int half = lane >> 4;
    const int l16  = lane & 15;
    const int h    = wave;
    const int bb   = blockIdx.x >> 4;
    const int q0   = (blockIdx.x & 15) * 16;

    if (t < S_) msk[t] = mask[bb * S_ + t] ? 1.0f : 0.0f;

    // Preload Q A-fragments for this head (rows q0..q0+15, dk 0..63).
    const bf16* qrow = qh + ((size_t)((bb * NH_ + h) * S_) + q0 + l16) * DK_;
    const v16bf qa0 = a_frag_bf16(qrow + 0, lane);
    const v16bf qa1 = a_frag_bf16(qrow + 32, lane);

    const bf16* khTh = khT + (size_t)(bb * NH_ + h) * DK_ * S_;
    const bf16* vhh  = vh  + (size_t)(bb * NH_ + h) * S_ * DK_;

    float rowm[8], rowl[8];
#pragma unroll
    for (int r = 0; r < 8; ++r) { rowm[r] = -1e30f; rowl[r] = 0.f; }
    v8f O0 = v8f_zero(), O1 = v8f_zero(), O2 = v8f_zero(), O3 = v8f_zero();

    __syncthreads();

    for (int it = 0; it < S_ / 32; ++it) {
        const int kb = it * 32;

        // ---- issue QK^T B-fragment loads early: latency hides under the ----
        // ---- geometry WMMA phase (compiler can't hoist past the barrier) ---
        const int kr = 16 * half + l16;
        const bf16* kp0 = khTh + (size_t)(kb + l16)      * DK_;  // keys 0..15
        const bf16* kp1 = khTh + (size_t)(kb + 16 + l16) * DK_;  // keys 16..31
        v16bf b00 = a_frag_bf16(kp0, lane);
        v16bf b01 = a_frag_bf16(kp1, lane);
        v16bf b10 = a_frag_bf16(kp0 + 32, lane);
        v16bf b11 = a_frag_bf16(kp1 + 32, lane);

        // ---- cooperative geometry bias: 32 pair-tiles, 4 per wave ----------
#pragma unroll
        for (int pp = 0; pp < 4; ++pp) {
            const int pt  = wave + pp * 8;
            const int qi  = pt >> 1;
            const int kh2 = pt & 1;                     // which 16-key half
            const float* prr = rr +
                ((size_t)(bb * S_ + q0 + qi) * S_ + kb + kh2 * 16 + l16) * GF_;
            if (it + 1 < S_ / 32)                        // global_prefetch_b8
                __builtin_prefetch(prr + (size_t)32 * GF_, 0, 1);
            v8f g = v8f_zero();
#pragma unroll
            for (int f0 = 0; f0 < GF_; f0 += 32) {
                v16bf a;
#pragma unroll
                for (int i = 0; i < 8; ++i) {
                    a[i]     = (bf16)prr[f0 + 8 * half + i];
                    a[i + 8] = (bf16)prr[f0 + 16 + 8 * half + i];
                }
                const int fr = f0 + 16 * half + l16;
                v16bf wb = a_frag_bf16(WgT + l16 * GF_ + f0, lane);
                g = wmma_bf16(a, wb, g);
            }
            const int head = l16;                        // D col = head
            if (head < NH_) {
                const float bgv = bg[head];
#pragma unroll
                for (int r = 0; r < 8; ++r) {
                    const int ki = kh2 * 16 + r + 8 * half;
                    // relu then clip(1e-6) then log == log(max(x, 1e-6))
                    geo[(qi * 32 + ki) * GSTR + head] = __logf(fmaxf(g[r] + bgv, 1e-6f));
                }
            }
        }
        __syncthreads();

        // ---- per-head QK^T over the 32-key slab ----------------------------
        v8f sA = v8f_zero(), sB = v8f_zero();
        sA = wmma_bf16(qa0, b00, sA); sA = wmma_bf16(qa1, b10, sA);
        sB = wmma_bf16(qa0, b01, sB); sB = wmma_bf16(qa1, b11, sB);

        // ---- issue att@V B-fragment loads early (consumed after softmax) ---
        const bf16* vrow = vhh + (size_t)l16 * S_ + kb;
        v16bf vb0 = a_frag_bf16(vrow, lane);
        v16bf vb1 = a_frag_bf16(vrow + 16 * S_, lane);
        v16bf vb2 = a_frag_bf16(vrow + 32 * S_, lane);
        v16bf vb3 = a_frag_bf16(vrow + 48 * S_, lane);

        // ---- masked online softmax over 32 columns -------------------------
#pragma unroll
        for (int r = 0; r < 8; ++r) {
            const int qi  = r + 8 * half;
            const int kiA = l16;
            const int kiB = 16 + l16;
            float scA = (msk[kb + kiA] != 0.f) ? -1e9f
                        : (sA[r] * 0.125f + geo[(qi * 32 + kiA) * GSTR + h]);
            float scB = (msk[kb + kiB] != 0.f) ? -1e9f
                        : (sB[r] * 0.125f + geo[(qi * 32 + kiB) * GSTR + h]);
            float mr = fmaxf(scA, scB);
#pragma unroll
            for (int off = 1; off < 16; off <<= 1)
                mr = fmaxf(mr, __shfl_xor(mr, off, 32));
            const float newm = fmaxf(rowm[r], mr);
            const float corr = __expf(rowm[r] - newm);
            const float ea = __expf(scA - newm);
            const float eb = __expf(scB - newm);
            float ps = ea + eb;
#pragma unroll
            for (int off = 1; off < 16; off <<= 1)
                ps += __shfl_xor(ps, off, 32);
            rowl[r] = rowl[r] * corr + ps;
            rowm[r] = newm;
            O0[r] *= corr; O1[r] *= corr; O2[r] *= corr; O3[r] *= corr;
            bf16* pp_ = Pbuf + (h * 16 + qi) * PROW;   // private to this wave
            pp_[kiA] = (bf16)ea;
            pp_[kiB] = (bf16)eb;
        }

        // ---- att @ V : A = P (16x32), B = vh slab, 4 d-tiles --------------
        {
            const bf16* prow = Pbuf + (h * 16 + l16) * PROW;
            v16bf pa = a_frag_bf16(prow, lane);      // same-wave LDS RAW: in-order
            O0 = wmma_bf16(pa, vb0, O0);
            O1 = wmma_bf16(pa, vb1, O1);
            O2 = wmma_bf16(pa, vb2, O2);
            O3 = wmma_bf16(pa, vb3, O3);
        }
        __syncthreads();   // geo reuse next slab
    }

    // ---- normalize and write attention output (bf16, row-major [B,S,H]) ---
#pragma unroll
    for (int r = 0; r < 8; ++r) {
        const float inv = 1.0f / rowl[r];
        O0[r] *= inv; O1[r] *= inv; O2[r] *= inv; O3[r] *= inv;
    }
#pragma unroll
    for (int r = 0; r < 8; ++r) {
        const int qi = r + 8 * half;
        bf16* op = attout + ((size_t)(bb * S_ + q0 + qi)) * H_ + h * DK_;
        op[0  + l16] = (bf16)O0[r];
        op[16 + l16] = (bf16)O1[r];
        op[32 + l16] = (bf16)O2[r];
        op[48 + l16] = (bf16)O3[r];
    }
}

// ---------------------------------------------------------------------------
// Kernel 4: output GEMM  out = attout @ Wm.T + bm   (fp32 result)
// ---------------------------------------------------------------------------
__global__ __launch_bounds__(256) void out_gemm_kernel(
    const bf16* __restrict__ attout, const bf16* __restrict__ WmT,
    const float* __restrict__ bm, float* __restrict__ out)
{
    __shared__ __align__(32) bf16 Abuf[16 * AROW];   // bank-skewed rows

    const int t    = threadIdx.x;
    const int wave = t >> 5;
    const int lane = t & 31;
    const int half = lane >> 4;
    const int l16  = lane & 15;
    const int r0   = blockIdx.x * 16;

    {
        const int row = t >> 4;
        const int c0  = (t & 15) * 32;
        const bf16* src = attout + (size_t)(r0 + row) * H_ + c0;
        bf16* dst = Abuf + row * AROW + c0;
#pragma unroll
        for (int i = 0; i < 32; i += 8)
            *(v8bf*)(dst + i) = *(const v8bf*)(src + i);
    }
    __syncthreads();

    const bf16* arow = Abuf + l16 * AROW;
#pragma unroll
    for (int nt = 0; nt < 4; ++nt) {
        const int n0 = (wave + nt * 8) * 16;
        v8f acc = v8f_zero();
        v16bf a_cur = a_frag_bf16(arow, lane);
        v16bf b_cur = a_frag_bf16(WmT + (size_t)(n0 + l16) * H_, lane);
#pragma unroll
        for (int k0 = 0; k0 < H_; k0 += 32) {
            v16bf a_nxt = a_cur, b_nxt = b_cur;
            if (k0 + 32 < H_) {
                a_nxt = a_frag_bf16(arow + k0 + 32, lane);
                b_nxt = a_frag_bf16(WmT + (size_t)(n0 + l16) * H_ + k0 + 32, lane);
            }
            acc = wmma_bf16(a_cur, b_cur, acc);
            a_cur = a_nxt; b_cur = b_nxt;
        }
        const int col = n0 + l16;
        const float bv = bm[col];
#pragma unroll
        for (int r = 0; r < 8; ++r)
            out[(size_t)(r0 + r + 8 * half) * H_ + col] = acc[r] + bv;
    }
}

// ---------------------------------------------------------------------------
// Host launcher
// ---------------------------------------------------------------------------
extern "C" void kernel_launch(void* const* d_in, const int* in_sizes, int n_in,
                              void* d_out, int out_size, void* d_ws, size_t ws_size,
                              hipStream_t stream) {
    (void)in_sizes; (void)n_in; (void)out_size; (void)ws_size;
    // input order from setup_inputs():
    // 0 v (ignored by the module), 1 k, 2 q, 3 c, 4 mask, 5 region_abs,
    // 6 region_rel, 7 Wk, 8 bk, 9 Wq, 10 bq, 11 wgk, 12 bgk, 13 wgq, 14 bgq,
    // 15 wc1, 16 bc1, 17 wc2, 18 bc2, 19 Wg, 20 bg, 21 Wm, 22 bm
    const float* k_in  = (const float*)d_in[1];
    const float* q_in  = (const float*)d_in[2];
    const float* c_in  = (const float*)d_in[3];
    const unsigned char* mask = (const unsigned char*)d_in[4];  // numpy bool
    const float* rabs  = (const float*)d_in[5];
    const float* rrel  = (const float*)d_in[6];
    const float* Wk    = (const float*)d_in[7];
    const float* bk    = (const float*)d_in[8];
    const float* Wq    = (const float*)d_in[9];
    const float* bq    = (const float*)d_in[10];
    const float* wgk   = (const float*)d_in[11];
    const float* bgk   = (const float*)d_in[12];
    const float* wgq   = (const float*)d_in[13];
    const float* bgq   = (const float*)d_in[14];
    const float* wc1   = (const float*)d_in[15];
    const float* bc1   = (const float*)d_in[16];
    const float* wc2   = (const float*)d_in[17];
    const float* bc2   = (const float*)d_in[18];
    const float* Wg    = (const float*)d_in[19];
    const float* bg    = (const float*)d_in[20];
    const float* Wm    = (const float*)d_in[21];
    const float* bm    = (const float*)d_in[22];

    // workspace carve-up (256B aligned slices; ~17.5 MB total)
    char* ws = (char*)d_ws;
    size_t off = 0;
    auto take = [&](size_t bytes) -> char* {
        char* p = ws + off;
        off += (bytes + 255) & ~(size_t)255;
        return p;
    };
    bf16* WkT    = (bf16*)take((size_t)H_ * H_ * sizeof(bf16));           // 512 KB
    bf16* WqT    = (bf16*)take((size_t)H_ * H_ * sizeof(bf16));           // 512 KB
    bf16* WmT    = (bf16*)take((size_t)H_ * H_ * sizeof(bf16));           // 512 KB
    bf16* WgT    = (bf16*)take((size_t)GF_ * 16 * sizeof(bf16));          //   3 KB
    bf16* qh     = (bf16*)take((size_t)B_ * NH_ * S_ * DK_ * sizeof(bf16)); // 4 MB
    bf16* khT    = (bf16*)take((size_t)B_ * NH_ * DK_ * S_ * sizeof(bf16)); // 4 MB
    bf16* vh     = (bf16*)take((size_t)B_ * NH_ * S_ * DK_ * sizeof(bf16)); // 4 MB
    bf16* attout = (bf16*)take((size_t)B_ * S_ * H_ * sizeof(bf16));       // 4 MB

    // 1. convert weights to bf16, same [n][k] layouts
    cvt_kernel<<<(H_ * H_) / 256, 256, 0, stream>>>(Wk, Wq, Wm, Wg,
                                                    WkT, WqT, WmT, WgT);
    // 2. projections + gating (k path writes transposed vh + kh; q path -> qh)
    const int row_blocks = (B_ * S_) / 16;   // 256
    proj_kernel<<<row_blocks, 256, 0, stream>>>(k_in, c_in, WkT, bk, wgk, bgk,
                                                wc1, bc1, rabs, khT, vh, 0);
    proj_kernel<<<row_blocks, 256, 0, stream>>>(q_in, c_in, WqT, bq, wgq, bgq,
                                                wc2, bc2, rabs, qh, vh, 1);
    // 3. fused geometry-bias flash attention (region_rel read exactly once)
    attn_kernel<<<B_ * (S_ / 16), 256, 0, stream>>>(qh, khT, vh, rrel, WgT, bg,
                                                    mask, attout);
    // 4. output projection to fp32
    out_gemm_kernel<<<row_blocks, 256, 0, stream>>>(attout, WmT, bm,
                                                    (float*)d_out);
}
